// GraphComponent_79456894976528
// MI455X (gfx1250) — matrix operand order repaired and checked by reading, not compiled, not run on hardware
//
#include <hip/hip_runtime.h>
#include <hip/hip_bf16.h>

typedef _Float16 h16 __attribute__((ext_vector_type(16)));
typedef _Float16 h8  __attribute__((ext_vector_type(8)));
typedef _Float16 h4  __attribute__((ext_vector_type(4)));
typedef float    f8  __attribute__((ext_vector_type(8)));

#define WMMA_F16(a, b, c) \
    __builtin_amdgcn_wmma_f32_16x16x32_f16(false, (a), false, (b), (short)0, (c), false, false)

__device__ __forceinline__ float gelu_exact(float x) {
    return 0.5f * x * (1.0f + erff(x * 0.70710678118654752440f));
}

__device__ __forceinline__ h16 cat16(h8 lo, h8 hi) {
    return __builtin_shufflevector(lo, hi, 0,1,2,3,4,5,6,7,8,9,10,11,12,13,14,15);
}

// A fragment: 16x32 f16, row-major source with leading dim lda (halfs).
// lane l: row = base + (l&15); halves 0-7 = K[kb..kb+7], 8-15 = K[kb+16..kb+23], kb = (l<16?0:8)
__device__ __forceinline__ h16 load_a_frag(const _Float16* A, int row, int lda, int ko, int lane) {
    const _Float16* p = A + (size_t)row * lda + ko + ((lane & 16) ? 8 : 0);
    h8 lo = *(const h8*)(p);
    h8 hi = *(const h8*)(p + 16);
    return cat16(lo, hi);
}

// B fragment from transposed weights [Nout][K] (f16, LDS). lane l: col = l&15,
// halves 0-7 = K[kb..kb+7], 8-15 = K[kb+16..kb+23], kb = ko + (l<16?0:8)
__device__ __forceinline__ h16 load_b_frag(const _Float16* sB, int ntile, int ldk, int ko, int lane) {
    const _Float16* p = sB + (size_t)(ntile * 16 + (lane & 15)) * ldk + ko + ((lane & 16) ? 8 : 0);
    h8 lo = *(const h8*)(p);
    h8 hi = *(const h8*)(p + 16);
    return cat16(lo, hi);
}

// ---- weight prep: fp32 row-major -> f16 transposed [Nout][K] -------------
__global__ void k_prep_weights(const float* __restrict__ Wc, const float* __restrict__ W1,
                               const float* __restrict__ W2, _Float16* __restrict__ WcT,
                               _Float16* __restrict__ W1T, _Float16* __restrict__ W2T) {
    int t = blockIdx.x * blockDim.x + threadIdx.x;
    if (t < 128 * 128) {
        int n = t >> 7, k = t & 127;
        WcT[n * 128 + k] = (_Float16)Wc[k * 128 + n];
    } else if (t < 128 * 128 + 512 * 128) {
        int i = t - 128 * 128;
        int n = i >> 7, k = i & 127;                 // n<512, k<128
        W1T[n * 128 + k] = (_Float16)W1[k * 512 + n];
    } else if (t < 128 * 128 + 512 * 128 + 128 * 512) {
        int i = t - 128 * 128 - 512 * 128;
        int n = i >> 9, k = i & 511;                 // n<128, k<512
        W2T[n * 512 + k] = (_Float16)W2[k * 128 + n];
    }
}

// ---- LN1 + exact GELU -> f16 (one wave per row, 4 elems/lane) ------------
__global__ void k_ln1_gelu(const float* __restrict__ x, const float* __restrict__ g,
                           const float* __restrict__ b, _Float16* __restrict__ out, int N) {
    int lane = threadIdx.x & 31;
    int rpb = blockDim.x >> 5;
    for (int row = blockIdx.x * rpb + (threadIdx.x >> 5); row < N; row += gridDim.x * rpb) {
        float4 v = *(const float4*)(x + (size_t)row * 128 + lane * 4);
        float s = v.x + v.y + v.z + v.w;
        for (int o = 16; o >= 1; o >>= 1) s += __shfl_xor(s, o, 32);
        float mu = s * (1.0f / 128.0f);
        float dx = v.x - mu, dy = v.y - mu, dz = v.z - mu, dw = v.w - mu;
        float q = dx * dx + dy * dy + dz * dz + dw * dw;
        for (int o = 16; o >= 1; o >>= 1) q += __shfl_xor(q, o, 32);
        float rstd = rsqrtf(q * (1.0f / 128.0f) + 1e-5f);
        int c = lane * 4;
        h4 o4;
        o4[0] = (_Float16)gelu_exact(dx * rstd * g[c + 0] + b[c + 0]);
        o4[1] = (_Float16)gelu_exact(dy * rstd * g[c + 1] + b[c + 1]);
        o4[2] = (_Float16)gelu_exact(dz * rstd * g[c + 2] + b[c + 2]);
        o4[3] = (_Float16)gelu_exact(dw * rstd * g[c + 3] + b[c + 3]);
        *(h4*)(out + (size_t)row * 128 + c) = o4;
    }
}

// ---- GEMM: h[N,128] @ Wc -> hW[N,128] fp32; WcT staged in 32KB LDS --------
__global__ void __launch_bounds__(256)
k_gemm_conv(const _Float16* __restrict__ A, const _Float16* __restrict__ WT,
            float* __restrict__ Cout, int N) {
    __shared__ _Float16 sW[128 * 128];
    {
        const uint4* src = (const uint4*)WT;
        uint4* dst = (uint4*)sW;
        for (int i = threadIdx.x; i < 2048; i += blockDim.x) dst[i] = src[i];
    }
    __syncthreads();
    int lane = threadIdx.x & 31;
    int wv = threadIdx.x >> 5;
    int nstrips = N >> 4;
    for (int strip = blockIdx.x * 8 + wv; strip < nstrips; strip += gridDim.x * 8) {
        int rowbase = strip << 4;
        int arow = rowbase + (lane & 15);
        h16 a0 = load_a_frag(A, arow, 128, 0, lane);
        h16 a1 = load_a_frag(A, arow, 128, 32, lane);
        h16 a2 = load_a_frag(A, arow, 128, 64, lane);
        h16 a3 = load_a_frag(A, arow, 128, 96, lane);
#pragma unroll
        for (int nt = 0; nt < 8; ++nt) {
            f8 acc = {};
            acc = WMMA_F16(a0, load_b_frag(sW, nt, 128, 0, lane), acc);
            acc = WMMA_F16(a1, load_b_frag(sW, nt, 128, 32, lane), acc);
            acc = WMMA_F16(a2, load_b_frag(sW, nt, 128, 64, lane), acc);
            acc = WMMA_F16(a3, load_b_frag(sW, nt, 128, 96, lane), acc);
            int col = nt * 16 + (lane & 15);
            int rofs = rowbase + ((lane & 16) ? 8 : 0);
#pragma unroll
            for (int i = 0; i < 8; ++i)
                Cout[(size_t)(rofs + i) * 128 + col] = acc[i];
        }
    }
}

// ---- degree / dinv / self-loop init --------------------------------------
__global__ void k_deg_init(float* __restrict__ deg, int N) {
    int i = blockIdx.x * blockDim.x + threadIdx.x;
    if (i < N) deg[i] = 1.0f;  // self-loop
}

__global__ void k_deg_scatter(const long long* __restrict__ ei, float* __restrict__ deg,
                              long long E) {
    long long e = (long long)blockIdx.x * blockDim.x + threadIdx.x;
    if (e < E) atomicAdd(&deg[ei[e + E]], 1.0f);  // dst row
}

__global__ void k_agg_init(const float* __restrict__ deg, float* __restrict__ dinv,
                           const float* __restrict__ hW, float* __restrict__ agg, int N) {
    int lane = threadIdx.x & 31;
    int rpb = blockDim.x >> 5;
    for (int row = blockIdx.x * rpb + (threadIdx.x >> 5); row < N; row += gridDim.x * rpb) {
        float di = rsqrtf(deg[row]);  // deg >= 1 always (self-loop)
        if (lane == 0) dinv[row] = di;
        float c = di * di;
        float4 v = *(const float4*)(hW + (size_t)row * 128 + lane * 4);
        v.x *= c; v.y *= c; v.z *= c; v.w *= c;
        *(float4*)(agg + (size_t)row * 128 + lane * 4) = v;
    }
}

// ---- edge scatter: one wave per edge, float4 per lane --------------------
__global__ void k_edge_scatter(const long long* __restrict__ ei, const float* __restrict__ dinv,
                               const float* __restrict__ hW, float* __restrict__ agg,
                               long long E) {
    long long t = (long long)blockIdx.x * blockDim.x + threadIdx.x;
    long long nt = (long long)gridDim.x * blockDim.x;
    long long total = E * 32;
    for (long long i = t; i < total; i += nt) {
        long long e = i >> 5;
        int lane = (int)(i & 31);
        long long s = ei[e], d = ei[e + E];
        float c = dinv[s] * dinv[d];
        float4 v = *(const float4*)(hW + s * 128 + lane * 4);
        float* ap = agg + d * 128 + lane * 4;
        atomicAdd(ap + 0, v.x * c);
        atomicAdd(ap + 1, v.y * c);
        atomicAdd(ap + 2, v.z * c);
        atomicAdd(ap + 3, v.w * c);
    }
}

// ---- conv residual + LN2 -> f16 ------------------------------------------
__global__ void k_conv_ln2(const float* __restrict__ nf, const float* __restrict__ agg,
                           const float* __restrict__ bc, const float* __restrict__ g,
                           const float* __restrict__ b, _Float16* __restrict__ out, int N) {
    int lane = threadIdx.x & 31;
    int rpb = blockDim.x >> 5;
    for (int row = blockIdx.x * rpb + (threadIdx.x >> 5); row < N; row += gridDim.x * rpb) {
        int c = lane * 4;
        float4 a = *(const float4*)(nf + (size_t)row * 128 + c);
        float4 v = *(const float4*)(agg + (size_t)row * 128 + c);
        v.x += a.x + bc[c + 0]; v.y += a.y + bc[c + 1];
        v.z += a.z + bc[c + 2]; v.w += a.w + bc[c + 3];
        float s = v.x + v.y + v.z + v.w;
        for (int o = 16; o >= 1; o >>= 1) s += __shfl_xor(s, o, 32);
        float mu = s * (1.0f / 128.0f);
        float dx = v.x - mu, dy = v.y - mu, dz = v.z - mu, dw = v.w - mu;
        float q = dx * dx + dy * dy + dz * dz + dw * dw;
        for (int o = 16; o >= 1; o >>= 1) q += __shfl_xor(q, o, 32);
        float rstd = rsqrtf(q * (1.0f / 128.0f) + 1e-5f);
        h4 o4;
        o4[0] = (_Float16)(dx * rstd * g[c + 0] + b[c + 0]);
        o4[1] = (_Float16)(dy * rstd * g[c + 1] + b[c + 1]);
        o4[2] = (_Float16)(dz * rstd * g[c + 2] + b[c + 2]);
        o4[3] = (_Float16)(dw * rstd * g[c + 3] + b[c + 3]);
        *(h4*)(out + (size_t)row * 128 + c) = o4;
    }
}

// ---- FFN1: f[N,128] @ W1 -> gelu -> t[N,512] f16; blockIdx.y = 128-col group
__global__ void __launch_bounds__(256)
k_gemm_ffn1(const _Float16* __restrict__ A, const _Float16* __restrict__ W1T,
            const float* __restrict__ b1, _Float16* __restrict__ T, int N) {
    __shared__ _Float16 sW[128 * 128];
    int ny = blockIdx.y;  // 0..3
    {
        const uint4* src = (const uint4*)(W1T + (size_t)ny * 128 * 128);
        uint4* dst = (uint4*)sW;
        for (int i = threadIdx.x; i < 2048; i += blockDim.x) dst[i] = src[i];
    }
    __syncthreads();
    int lane = threadIdx.x & 31;
    int wv = threadIdx.x >> 5;
    int nstrips = N >> 4;
    for (int strip = blockIdx.x * 8 + wv; strip < nstrips; strip += gridDim.x * 8) {
        int rowbase = strip << 4;
        int arow = rowbase + (lane & 15);
        h16 a0 = load_a_frag(A, arow, 128, 0, lane);
        h16 a1 = load_a_frag(A, arow, 128, 32, lane);
        h16 a2 = load_a_frag(A, arow, 128, 64, lane);
        h16 a3 = load_a_frag(A, arow, 128, 96, lane);
#pragma unroll
        for (int nt = 0; nt < 8; ++nt) {
            f8 acc = {};
            acc = WMMA_F16(a0, load_b_frag(sW, nt, 128, 0, lane), acc);
            acc = WMMA_F16(a1, load_b_frag(sW, nt, 128, 32, lane), acc);
            acc = WMMA_F16(a2, load_b_frag(sW, nt, 128, 64, lane), acc);
            acc = WMMA_F16(a3, load_b_frag(sW, nt, 128, 96, lane), acc);
            int col = ny * 128 + nt * 16 + (lane & 15);
            float bb = b1[col];
            int rofs = rowbase + ((lane & 16) ? 8 : 0);
#pragma unroll
            for (int i = 0; i < 8; ++i)
                T[(size_t)(rofs + i) * 512 + col] = (_Float16)gelu_exact(acc[i] + bb);
        }
    }
}

// ---- FFN2: t[N,512] @ W2 -> gelu -> +node_feat -> out fp32 ----------------
__global__ void __launch_bounds__(256)
k_gemm_ffn2(const _Float16* __restrict__ T, const _Float16* __restrict__ W2T,
            const float* __restrict__ b2, const float* __restrict__ nf,
            float* __restrict__ Out, int N) {
    __shared__ _Float16 sW[128 * 128];  // one 128-K chunk of W2T at a time
    int lane = threadIdx.x & 31;
    int wv = threadIdx.x >> 5;
    int nstrips = N >> 4;
    for (int strip0 = blockIdx.x * 8; strip0 < nstrips; strip0 += gridDim.x * 8) {
        int strip = strip0 + wv;
        bool valid = strip < nstrips;
        int rowbase = strip << 4;
        int arow = valid ? rowbase + (lane & 15) : (lane & 15);
        f8 zero = {};
        f8 acc[8];
#pragma unroll
        for (int nt = 0; nt < 8; ++nt) acc[nt] = zero;
        for (int kc = 0; kc < 4; ++kc) {
            __syncthreads();
            for (int i = threadIdx.x; i < 2048; i += blockDim.x) {
                int n = i >> 4, k16 = i & 15;
                ((uint4*)sW)[i] = ((const uint4*)(W2T + (size_t)n * 512 + kc * 128))[k16];
            }
            __syncthreads();
            if (valid) {
                int ko = kc * 128;
                h16 a0 = load_a_frag(T, arow, 512, ko, lane);
                h16 a1 = load_a_frag(T, arow, 512, ko + 32, lane);
                h16 a2 = load_a_frag(T, arow, 512, ko + 64, lane);
                h16 a3 = load_a_frag(T, arow, 512, ko + 96, lane);
#pragma unroll
                for (int nt = 0; nt < 8; ++nt) {
                    acc[nt] = WMMA_F16(a0, load_b_frag(sW, nt, 128, 0, lane), acc[nt]);
                    acc[nt] = WMMA_F16(a1, load_b_frag(sW, nt, 128, 32, lane), acc[nt]);
                    acc[nt] = WMMA_F16(a2, load_b_frag(sW, nt, 128, 64, lane), acc[nt]);
                    acc[nt] = WMMA_F16(a3, load_b_frag(sW, nt, 128, 96, lane), acc[nt]);
                }
            }
        }
        if (valid) {
            int rofs = rowbase + ((lane & 16) ? 8 : 0);
#pragma unroll
            for (int nt = 0; nt < 8; ++nt) {
                int col = nt * 16 + (lane & 15);
                float bb = b2[col];
#pragma unroll
                for (int i = 0; i < 8; ++i) {
                    size_t idx = (size_t)(rofs + i) * 128 + col;
                    Out[idx] = nf[idx] + gelu_exact(acc[nt][i] + bb);
                }
            }
        }
    }
}

extern "C" void kernel_launch(void* const* d_in, const int* in_sizes, int n_in,
                              void* d_out, int out_size, void* d_ws, size_t ws_size,
                              hipStream_t stream) {
    const float* node_feat = (const float*)d_in[0];
    const long long* edge_index = (const long long*)d_in[1];  // int64 [2,E]
    const float* ln1_g = (const float*)d_in[3];
    const float* ln1_b = (const float*)d_in[4];
    const float* W_conv = (const float*)d_in[5];
    const float* b_conv = (const float*)d_in[6];
    const float* ln2_g = (const float*)d_in[7];
    const float* ln2_b = (const float*)d_in[8];
    const float* W1 = (const float*)d_in[9];
    const float* b1 = (const float*)d_in[10];
    const float* W2 = (const float*)d_in[11];
    const float* b2 = (const float*)d_in[12];
    float* out = (float*)d_out;

    int N = in_sizes[0] / 128;            // 100000
    long long E = in_sizes[1] / 2;        // 1000000

    // workspace layout (bytes); t_f16 aliases hW+agg (both dead by then),
    // f_f16 reuses h_f16
    char* ws = (char*)d_ws;
    size_t szrow = (size_t)N * 128;
    float* hW = (float*)ws;                                   // szrow*4
    float* agg = (float*)(ws + szrow * 4);                    // szrow*4
    _Float16* hbuf = (_Float16*)(ws + szrow * 8);             // szrow*2 (h, then f)
    float* deg = (float*)(ws + szrow * 10);                   // N*4
    float* dinv = deg + N;                                    // N*4
    _Float16* WcT = (_Float16*)((char*)(dinv + N));           // 128*128 f16
    _Float16* W1T = WcT + 128 * 128;                          // 512*128 f16
    _Float16* W2T = W1T + 512 * 128;                          // 128*512 f16
    _Float16* Tbuf = (_Float16*)ws;                           // N*512 f16 (aliases hW+agg)

    int nstrips = N >> 4;
    int gblocks = (nstrips + 7) / 8;      // one wave per 16-row strip

    k_prep_weights<<<(128 * 128 + 512 * 128 + 128 * 512 + 255) / 256, 256, 0, stream>>>(
        W_conv, W1, W2, WcT, W1T, W2T);
    k_ln1_gelu<<<(N + 7) / 8, 256, 0, stream>>>(node_feat, ln1_g, ln1_b, hbuf, N);
    k_gemm_conv<<<gblocks, 256, 0, stream>>>(hbuf, WcT, hW, N);
    k_deg_init<<<(N + 255) / 256, 256, 0, stream>>>(deg, N);
    k_deg_scatter<<<(int)((E + 255) / 256), 256, 0, stream>>>(edge_index, deg, E);
    k_agg_init<<<(N + 7) / 8, 256, 0, stream>>>(deg, dinv, hW, agg, N);
    long long sc_threads = E * 32;
    int sc_blocks = (int)((sc_threads + 255) / 256);
    k_edge_scatter<<<sc_blocks, 256, 0, stream>>>(edge_index, dinv, hW, agg, E);
    k_conv_ln2<<<(N + 7) / 8, 256, 0, stream>>>(node_feat, agg, b_conv, ln2_g, ln2_b, hbuf, N);
    k_gemm_ffn1<<<dim3(gblocks, 4), 256, 0, stream>>>(hbuf, W1T, b1, Tbuf, N);
    k_gemm_ffn2<<<gblocks, 256, 0, stream>>>(Tbuf, W2T, b2, node_feat, out, N);

    (void)n_in; (void)out_size; (void)ws_size; (void)d_in;
}
